// MultiHeadAttention_31267361915281
// MI455X (gfx1250) — matrix operand order repaired and checked by reading, not compiled, not run on hardware
//
#include <hip/hip_runtime.h>
#include <hip/hip_bf16.h>
#include <math.h>

typedef __bf16 bf16_t;
typedef __attribute__((ext_vector_type(16))) __bf16 v16bf;
typedef __attribute__((ext_vector_type(8)))  __bf16 v8bf;
typedef __attribute__((ext_vector_type(8)))  float  v8f;
typedef __attribute__((ext_vector_type(8)))  unsigned int v8u;
typedef int v4i __attribute__((vector_size(16)));   // matches builtin param type

#define H_ 16
#define M_ 1024
#define KD 64
#define VD 64
#define B_ 2
#define C_ 2048

#define AS1 __attribute__((address_space(1)))
#define AS3 __attribute__((address_space(3)))

#if __has_builtin(__builtin_amdgcn_global_load_async_to_lds_b128) && \
    __has_builtin(__builtin_amdgcn_s_wait_asynccnt)
#define USE_ASYNC_LDS 1
#endif

#ifdef USE_ASYNC_LDS
// Per-lane async DMA: LDS[l] = MEM[g] (16 bytes), tracked by ASYNCcnt.
__device__ __forceinline__ void async_cp_b128(const void* g, void* l) {
  __builtin_amdgcn_global_load_async_to_lds_b128(
      (AS1 v4i*)(AS1 void*)(void*)g, (AS3 v4i*)(AS3 void*)l, 0, 0);
}
__device__ __forceinline__ void async_wait0() {
  __builtin_amdgcn_s_wait_asynccnt(0);
}
#endif

// D = A(16x32 bf16) * B(32x16 bf16) + C(16x16 f32)
__device__ __forceinline__ v8f wmma_bf16(v16bf a, v16bf b, v8f c) {
  return __builtin_amdgcn_wmma_f32_16x16x32_bf16(false, a, false, b, (short)0, c,
                                                 false, false);
}

__device__ __forceinline__ v16bf cat8(v8bf lo, v8bf hi) {
  return __builtin_shufflevector(lo, hi, 0, 1, 2, 3, 4, 5, 6, 7,
                                 8, 9, 10, 11, 12, 13, 14, 15);
}

// A fragment (16x32 bf16) from LDS row-major tile -> two ds_load_b128.
__device__ __forceinline__ v16bf frag_a_lds(const bf16_t* p, int stride) {
  const int lane = threadIdx.x & 31;
  const bf16_t* r = p + (lane & 15) * stride + ((lane >> 4) << 4);
  return cat8(*(const v8bf*)r, *(const v8bf*)(r + 8));
}

// B fragment (32x16 bf16) from a PAIR-PACKED u32 tile: unit row kk holds
// (k=2kk, k=2kk+1) for every column. 8 direct b32 LDS loads, no repacking.
__device__ __forceinline__ v16bf frag_b_pk(const unsigned int* p, int stride) {
  const int lane = threadIdx.x & 31;
  const int n   = lane & 15;
  const int kk0 = (lane >> 4) << 3;
  v8u x;
#pragma unroll
  for (int j = 0; j < 8; ++j) x[j] = p[(kk0 + j) * stride + n];
  return __builtin_bit_cast(v16bf, x);
}

__device__ __forceinline__ unsigned int pk_f32(float lo, float hi) {
  const unsigned short l = __builtin_bit_cast(unsigned short, (bf16_t)lo);
  const unsigned short h = __builtin_bit_cast(unsigned short, (bf16_t)hi);
  return (unsigned int)l | ((unsigned int)h << 16);
}
__device__ __forceinline__ unsigned int pk_bf(bf16_t lo, bf16_t hi) {
  const unsigned short l = __builtin_bit_cast(unsigned short, lo);
  const unsigned short h = __builtin_bit_cast(unsigned short, hi);
  return (unsigned int)l | ((unsigned int)h << 16);
}

// -------------------- Kernel 1: QKV projection --------------------
__global__ __launch_bounds__(256)
void qkv_proj_kernel(const float* __restrict__ qin, const float* __restrict__ kvin,
                     const float* __restrict__ wq, const float* __restrict__ wk,
                     const float* __restrict__ wv, bf16_t* __restrict__ ws) {
  const int which = blockIdx.z;              // 0=Q 1=K 2=V
  const int bh = blockIdx.y;
  const int b  = bh / H_;
  const int h  = bh % H_;
  const int c0 = blockIdx.x * 128;

  const float* in = (which == 0) ? qin : kvin;
  const float* w  = (which == 0) ? wq : (which == 1 ? wk : wv);
  bf16_t* outp = ws + (size_t)which * ((size_t)B_ * H_ * C_ * KD);

  __shared__ bf16_t       sA[128][40];   // input tile, row-major (A operand)
  __shared__ unsigned int sWp[16][68];   // weight tile, pair-packed (B operand)

  const int tid  = threadIdx.x;
  const int lane = tid & 31;
  const int wid  = tid >> 5;

  v8f acc[4];
#pragma unroll
  for (int t = 0; t < 4; ++t)
#pragma unroll
    for (int e = 0; e < 8; ++e) acc[t][e] = 0.0f;

  for (int m0 = 0; m0 < M_; m0 += 32) {
    { // A: 128x32 fp32 -> bf16 (packed b64 stores)
      const int r0 = tid >> 3;
      const int cc = (tid & 7) * 4;
#pragma unroll
      for (int rr = 0; rr < 4; ++rr) {
        const int r = r0 + rr * 32;
        const float4 f = *(const float4*)(in + ((size_t)b * C_ + c0 + r) * M_ + m0 + cc);
        uint2 u;
        u.x = pk_f32(f.x, f.y);
        u.y = pk_f32(f.z, f.w);
        *(uint2*)&sA[r][cc] = u;
      }
    }
    { // W: 32x64 fp32 -> pair-packed u32 (one b128 store per thread)
      const int mp = tid >> 4;          // pair-row 0..15
      const int c4 = (tid & 15) * 4;
      const float* wrow = w + ((size_t)h * M_ + m0 + 2 * mp) * KD + c4;
      const float4 f0 = *(const float4*)wrow;
      const float4 f1 = *(const float4*)(wrow + KD);
      uint4 u;
      u.x = pk_f32(f0.x, f1.x); u.y = pk_f32(f0.y, f1.y);
      u.z = pk_f32(f0.z, f1.z); u.w = pk_f32(f0.w, f1.w);
      *(uint4*)&sWp[mp][c4] = u;
    }
    __syncthreads();
    const v16bf a = frag_a_lds(&sA[wid * 16][0], 40);
#pragma unroll
    for (int t = 0; t < 4; ++t) {
      const v16bf bb = frag_b_pk(&sWp[0][t * 16], 68);
      acc[t] = wmma_bf16(a, bb, acc[t]);
    }
    __syncthreads();
  }
  const int n   = lane & 15;
  const int kh8 = (lane >> 4) << 3;
#pragma unroll
  for (int t = 0; t < 4; ++t)
#pragma unroll
    for (int i = 0; i < 8; ++i)
      outp[((size_t)bh * C_ + c0 + wid * 16 + kh8 + i) * KD + t * 16 + n] =
          (bf16_t)acc[t][i];
}

// -------------------- Kernel 2: flash attention --------------------
__global__ __launch_bounds__(256)
void attn_kernel(const bf16_t* __restrict__ ws, bf16_t* __restrict__ pre) {
  const int bh    = blockIdx.y;
  const int dTile = blockIdx.x * 128;
  const int tid   = threadIdx.x;
  const int lane  = tid & 31;
  const int wid   = tid >> 5;
  const int n     = lane & 15;
  const int kh8   = (lane >> 4) << 3;
  const int kh16  = (lane >> 4) << 4;

  const size_t QKV = (size_t)B_ * H_ * C_ * KD;
  const bf16_t* Qp = ws;
  const bf16_t* Kp = ws + QKV;
  const bf16_t* Vp = ws + 2 * QKV;

  __shared__ bf16_t       sK[64][72];      // K tile row-major (A operand)
  __shared__ unsigned int sVp[32][68];     // V tile pair-packed (B operand)
  __shared__ bf16_t       sP[8][16][72];   // per-wave P^T block [d][c] (A operand)

  const int dBase = dTile + wid * 16;

  // Q as two B-fragments (k chunks 0..31 / 32..63), register-resident.
  v16bf qb[2];
#pragma unroll
  for (int kc = 0; kc < 2; ++kc) {
    const bf16_t* qrow = Qp + ((size_t)bh * C_ + dBase + n) * KD + kc * 32 + kh16;
    qb[kc] = *(const v16bf*)qrow;
  }

  float mRun = -3.0e38f, lRun = 0.0f;
  v8f acc[4], zacc;
#pragma unroll
  for (int e = 0; e < 8; ++e) zacc[e] = 0.0f;
#pragma unroll
  for (int t = 0; t < 4; ++t) acc[t] = zacc;

  const float C1 = 0.18033688011112042f;   // (1/sqrt(K)) * log2(e)
  bf16_t* pw = &sP[wid][0][0];

  for (int cBlk = 0; cBlk < C_; cBlk += 64) {
    { // K tile: 64x64 bf16, row-major (async DMA to LDS when available)
      const int r  = tid >> 2;
      const int cc = (tid & 3) * 16;
      const bf16_t* krow = Kp + ((size_t)bh * C_ + cBlk + r) * KD + cc;
#ifdef USE_ASYNC_LDS
      async_cp_b128(krow,     &sK[r][cc]);
      async_cp_b128(krow + 8, &sK[r][cc + 8]);
#else
      *(v8bf*)&sK[r][cc]     = *(const v8bf*)krow;
      *(v8bf*)&sK[r][cc + 8] = *(const v8bf*)(krow + 8);
#endif
      if (cBlk + 64 < C_) __builtin_prefetch(krow + (size_t)64 * KD, 0, 0);
    }
    { // V tile: pair-pack two c-rows into u32 units (needs VALU, stays sync)
      const int cp = tid >> 3;          // pair-row 0..31
      const int c8 = (tid & 7) * 8;
      const bf16_t* vrow = Vp + ((size_t)bh * C_ + cBlk + 2 * cp) * VD + c8;
      const v8bf e = *(const v8bf*)vrow;
      const v8bf o = *(const v8bf*)(vrow + VD);
      uint4 u0, u1;
      u0.x = pk_bf(e[0], o[0]); u0.y = pk_bf(e[1], o[1]);
      u0.z = pk_bf(e[2], o[2]); u0.w = pk_bf(e[3], o[3]);
      u1.x = pk_bf(e[4], o[4]); u1.y = pk_bf(e[5], o[5]);
      u1.z = pk_bf(e[6], o[6]); u1.w = pk_bf(e[7], o[7]);
      *(uint4*)&sVp[cp][c8]     = u0;
      *(uint4*)&sVp[cp][c8 + 4] = u1;
      if (cBlk + 64 < C_) __builtin_prefetch(vrow + (size_t)64 * VD, 0, 0);
    }
#ifdef USE_ASYNC_LDS
    async_wait0();                       // own async copies landed in LDS
#endif
    __syncthreads();

    // S[c,d] for 4 c-subtiles: A = K tile rows, B = Q fragments
    v8f s[4];
#pragma unroll
    for (int cb = 0; cb < 4; ++cb) {
      const v16bf a0 = frag_a_lds(&sK[cb * 16][0], 72);
      const v16bf a1 = frag_a_lds(&sK[cb * 16][32], 72);
      v8f z = wmma_bf16(a0, qb[0], zacc);
      z = wmma_bf16(a1, qb[1], z);
      s[cb] = z;
    }

    // mask (applied before scaling, as in reference) + block max over c
    const int dG = dBase + n;
    float bm = -3.0e38f;
#pragma unroll
    for (int cb = 0; cb < 4; ++cb)
#pragma unroll
      for (int i = 0; i < 8; ++i) {
        const int cG = cBlk + cb * 16 + kh8 + i;
        float sv = s[cb][i];
        if (dG >= cG) sv -= 100.0f;      // triu(-100) on [c,d]
        s[cb][i] = sv;
        bm = fmaxf(bm, sv);
      }
    bm = fmaxf(bm, __shfl_xor(bm, 16, 32));
    const float mNew  = fmaxf(mRun, bm);
    const float alpha = exp2f((mRun - mNew) * C1);

    // P = exp; spill P^T (row d, contiguous c) with one b128 store per subtile
    float psum = 0.0f;
#pragma unroll
    for (int cb = 0; cb < 4; ++cb) {
      v8bf pv;
#pragma unroll
      for (int i = 0; i < 8; ++i) {
        const float p = exp2f((s[cb][i] - mNew) * C1);
        psum += p;
        pv[i] = (bf16_t)p;
      }
      *(v8bf*)&pw[n * 72 + cb * 16 + kh8] = pv;
    }
    psum += __shfl_xor(psum, 16, 32);
    lRun = lRun * alpha + psum;
    mRun = mNew;

    // rescale accumulator rows by alpha (redistribute per output row d)
    float aRow[8];
#pragma unroll
    for (int i = 0; i < 8; ++i) aRow[i] = __shfl(alpha, kh8 + i, 32);
#pragma unroll
    for (int t = 0; t < 4; ++t)
#pragma unroll
      for (int i = 0; i < 8; ++i) acc[t][i] *= aRow[i];

    // acc(d,v) += P^T(16d x 64c) * V(64c x 64v)
#pragma unroll
    for (int cc = 0; cc < 2; ++cc) {
      const v16bf pa = frag_a_lds(&pw[cc * 32], 72);
#pragma unroll
      for (int t = 0; t < 4; ++t) {
        const v16bf vb = frag_b_pk(&sVp[cc * 16][t * 16], 68);
        acc[t] = wmma_bf16(pa, vb, acc[t]);
      }
    }
    __syncthreads();
  }

  // normalize by row sums, store pre as bf16
  float rcp[8];
#pragma unroll
  for (int i = 0; i < 8; ++i) {
    const float lr = __shfl(lRun, kh8 + i, 32);
    rcp[i] = 1.0f / lr;
  }
#pragma unroll
  for (int t = 0; t < 4; ++t)
#pragma unroll
    for (int i = 0; i < 8; ++i)
      pre[((size_t)bh * C_ + dBase + kh8 + i) * VD + t * 16 + n] =
          (bf16_t)(acc[t][i] * rcp[i]);
}

// -------------------- Kernel 3: output projection --------------------
__global__ __launch_bounds__(256)
void out_proj_kernel(const bf16_t* __restrict__ pre, const float* __restrict__ wo,
                     float* __restrict__ out) {
  const int b  = blockIdx.z;
  const int d0 = blockIdx.y * 128;
  const int m0 = blockIdx.x * 64;
  const int tid  = threadIdx.x;
  const int lane = tid & 31;
  const int wid  = tid >> 5;
  const int n    = lane & 15;
  const int kh8  = (lane >> 4) << 3;

  __shared__ bf16_t       sA[128][40];
  __shared__ unsigned int sWp[16][68];

  v8f acc[4];
#pragma unroll
  for (int t = 0; t < 4; ++t)
#pragma unroll
    for (int e = 0; e < 8; ++e) acc[t][e] = 0.0f;

  for (int p0 = 0; p0 < H_ * VD; p0 += 32) {
    const int h  = p0 / VD;
    const int vb = p0 % VD;
    { // A: 128x32 bf16 straight copy (async DMA to LDS when available)
      const int r  = tid >> 1;
      const int cc = (tid & 1) * 16;
      const bf16_t* arow = pre + (((size_t)b * H_ + h) * C_ + d0 + r) * VD + vb + cc;
#ifdef USE_ASYNC_LDS
      async_cp_b128(arow,     &sA[r][cc]);
      async_cp_b128(arow + 8, &sA[r][cc + 8]);
#else
      *(v8bf*)&sA[r][cc]     = *(const v8bf*)arow;
      *(v8bf*)&sA[r][cc + 8] = *(const v8bf*)(arow + 8);
#endif
    }
    { // W: 32x64 fp32 -> pair-packed u32
      const int mp = tid >> 4;
      const int c4 = (tid & 15) * 4;
      const float* wrow = wo + ((size_t)h * VD + vb + 2 * mp) * M_ + m0 + c4;
      const float4 f0 = *(const float4*)wrow;
      const float4 f1 = *(const float4*)(wrow + M_);
      uint4 u;
      u.x = pk_f32(f0.x, f1.x); u.y = pk_f32(f0.y, f1.y);
      u.z = pk_f32(f0.z, f1.z); u.w = pk_f32(f0.w, f1.w);
      *(uint4*)&sWp[mp][c4] = u;
    }
#ifdef USE_ASYNC_LDS
    async_wait0();
#endif
    __syncthreads();
    const v16bf a = frag_a_lds(&sA[wid * 16][0], 40);
#pragma unroll
    for (int t = 0; t < 4; ++t) {
      const v16bf bb = frag_b_pk(&sWp[0][t * 16], 68);
      acc[t] = wmma_bf16(a, bb, acc[t]);
    }
    __syncthreads();
  }
#pragma unroll
  for (int t = 0; t < 4; ++t)
#pragma unroll
    for (int i = 0; i < 8; ++i)
      out[((size_t)b * C_ + d0 + wid * 16 + kh8 + i) * M_ + m0 + t * 16 + n] =
          acc[t][i];
}

extern "C" void kernel_launch(void* const* d_in, const int* in_sizes, int n_in,
                              void* d_out, int out_size, void* d_ws, size_t ws_size,
                              hipStream_t stream) {
  (void)in_sizes; (void)n_in; (void)out_size; (void)ws_size;
  const float* kvin = (const float*)d_in[0];
  const float* qin  = (const float*)d_in[1];
  const float* wq   = (const float*)d_in[2];
  const float* wk   = (const float*)d_in[3];
  const float* wv   = (const float*)d_in[4];
  const float* wo   = (const float*)d_in[5];
  float* out = (float*)d_out;

  bf16_t* ws = (bf16_t*)d_ws;
  const size_t QKV = (size_t)B_ * H_ * C_ * KD;   // elements per tensor
  bf16_t* preB = ws + 3 * QKV;                    // total ws need: 32 MB

  qkv_proj_kernel<<<dim3(C_ / 128, B_ * H_, 3), 256, 0, stream>>>(
      qin, kvin, wq, wk, wv, ws);
  attn_kernel<<<dim3(C_ / 128, B_ * H_), 256, 0, stream>>>(ws, preB);
  out_proj_kernel<<<dim3(M_ / 64, C_ / 128, B_), 256, 0, stream>>>(preB, wo, out);
}
